// LNN_16655883174535
// MI455X (gfx1250) — matrix operand order repaired and checked by reading, not compiled, not run on hardware
//
#include <hip/hip_runtime.h>
#include <cmath>

typedef __attribute__((ext_vector_type(16))) _Float16 v16h;
typedef __attribute__((ext_vector_type(8)))  _Float16 v8h;
typedef __attribute__((ext_vector_type(8)))  float    v8f;

#define NDOF 12
#define HID  256
#define NS   16     // samples per tile
#define NCOL 256    // NS * 16 tangent columns (13 used + grad col 13 + pad)

// ---------------- ws static region (offsets in halfs) ----------------
#define S_WH(k)  ((size_t)(k) * 65536)                    // W1..W3 f16 row-major (256x256)
#define S_WT(k)  ((size_t)196608 + (size_t)(k) * 65536)   // W1^T..W3^T f16 row-major
#define S_WH0    ((size_t)393216)                         // W0 f16, 256x32 (K padded 24->32)
#define S_WT0    ((size_t)401408)                         // W0^T f16, 32x256 (M padded)
#define S_W4     ((size_t)409600)                         // w4 f16, 256
#define S_HALFS  ((size_t)409856)
#define S_BYTES  ((size_t)((S_HALFS * 2 + 4095) & ~(size_t)4095))

// ---------------- per-block region (offsets in halfs) ----------------
#define B_T(k)   ((size_t)(k) * 65536)                    // T1..T4, col-major [col][256]
#define B_SE     ((size_t)262144)                         // extra S buffer
#define B_SIG(k) ((size_t)327680 + (size_t)(k) * 4096)    // sigma_k [sample][256] f16
#define B_CC(k)  ((size_t)344064 + (size_t)(k) * 4096)    // c_k     [sample][256] f16
#define B_UA     ((size_t)360448)                         // u4 (preserved for backward)
#define B_UB     ((size_t)364544)
#define B_UC     ((size_t)368640)
#define B_ZA     ((size_t)372736)                         // activations ping
#define B_ZB     ((size_t)376832)                         // activations pong
#define B_V      ((size_t)380928)                         // input tangents, [col][32]
#define B_Z0     ((size_t)389120)                         // network input, [sample][32]
#define B_HALFS  ((size_t)390144)
#define B_HVOFF  (B_HALFS * 2)                            // byte offset of f32 HV [256 col][32 row]
#define B_BYTES  ((size_t)((B_HALFS * 2 + 32768 + 4095) & ~(size_t)4095))

#define WMMA16(a, b, c) \
  __builtin_amdgcn_wmma_f32_16x16x32_f16(false, (a), false, (b), (short)0, (c), false, false)

// ---- fragment loaders for wave32 WMMA; A and B share one contiguous k order (not the ISA table order) ----
// A 16x32 f16: lane m = lane&15; half = lane>>4; elems e -> K = k0+16*half+e
// (one contiguous 32B chunk, row-major A).
__device__ __forceinline__ v16h loadA(const _Float16* __restrict__ A, int lda, int m0, int k0,
                                      int lane) {
  int m  = m0 + (lane & 15);
  int h16 = (lane >> 4) * 16;
  return *(const v16h*)(A + (size_t)m * lda + k0 + h16);
}
// B 32x16 f16: lane n = lane&15; elems e -> K = k0 + 16*half + e (one 32B chunk,
// B stored column-major [n][k], k contiguous).
__device__ __forceinline__ v16h loadB(const _Float16* __restrict__ Bc, int ldb, int n0, int k0,
                                      int lane) {
  int n   = n0 + (lane & 15);
  int h16 = (lane >> 4) * 16;
  return *(const v16h*)(Bc + (size_t)n * ldb + k0 + h16);
}

// workgroup-scope fence: all cross-wave traffic stays in this WGP's cache level
__device__ __forceinline__ void gsync() { __threadfence_block(); __syncthreads(); }

// shared-exp branchless sigmoid + softplus:
//   e = exp(-|a|); r = rcp(1+e); sig = a>=0 ? r : e*r; sp = max(a,0) + log(1+e)
__device__ __forceinline__ void sig_softplus(float a, float& sig, float& sp) {
  float e  = __expf(-fabsf(a));
  float d  = 1.0f + e;
  float r  = __builtin_amdgcn_rcpf(d);
  sig = (a >= 0.0f) ? r : e * r;
  sp  = fmaxf(a, 0.0f) + __logf(d);
}

// ---------------- forward: a = W z + b ; sigma, softplus / last-layer c4,u4 ----
template <int K, int LDA, int LDZ>
__device__ __forceinline__ void fwd_gemm(const _Float16* __restrict__ W,
                                         const _Float16* __restrict__ zin,
                                         const float* __restrict__ bias,
                                         _Float16* __restrict__ sig, _Float16* __restrict__ zout,
                                         const _Float16* __restrict__ w4,
                                         _Float16* __restrict__ c4, _Float16* __restrict__ u4,
                                         int wave, int lane) {
  for (int t = wave; t < 16; t += 8) {
    int m0 = t << 4;
    v8f acc = {};
#pragma unroll
    for (int kb = 0; kb < K; kb += 32) {
      v16h a = loadA(W, LDA, m0, kb, lane);
      v16h b = loadB(zin, LDZ, 0, kb, lane);
      acc = WMMA16(a, b, acc);
    }
    int smp = lane & 15;
    int mb  = m0 + (lane >> 4) * 8;
    v8f bv = *(const v8f*)(bias + mb);          // one 32B vector load for the bias chunk
    v8h so, zo, co, uo;
#pragma unroll
    for (int r = 0; r < 8; r++) {
      float av = acc[r] + bv[r];
      float sg, sp;
      sig_softplus(av, sg, sp);
      so[r] = (_Float16)sg;
      zo[r] = (_Float16)sp;
      float w = w4 ? (float)w4[mb + r] : 0.0f;
      co[r] = (_Float16)(w * sg * (1.0f - sg));
      uo[r] = (_Float16)(w * sg);
    }
    *(v8h*)(sig + smp * 256 + mb) = so;
    if (zout) *(v8h*)(zout + smp * 256 + mb) = zo;
    if (c4) { *(v8h*)(c4 + smp * 256 + mb) = co; *(v8h*)(u4 + smp * 256 + mb) = uo; }
  }
}

// ---------------- gradient backward: g = W^T u ; c = g*s*(1-s) ; u = g*s ----
__device__ __forceinline__ void grad_gemm(const _Float16* __restrict__ Wt,
                                          const _Float16* __restrict__ uin,
                                          const _Float16* __restrict__ sig,
                                          _Float16* __restrict__ cc,
                                          _Float16* __restrict__ uout, int wave, int lane) {
  for (int t = wave; t < 16; t += 8) {
    int m0 = t << 4;
    v8f acc = {};
#pragma unroll
    for (int kb = 0; kb < 256; kb += 32) {
      v16h a = loadA(Wt, 256, m0, kb, lane);
      v16h b = loadB(uin, 256, 0, kb, lane);
      acc = WMMA16(a, b, acc);
    }
    int smp = lane & 15;
    int mb  = m0 + (lane >> 4) * 8;
    v8h sv = *(const v8h*)(sig + smp * 256 + mb);
    v8h co, uo;
#pragma unroll
    for (int r = 0; r < 8; r++) {
      float g = acc[r], s = (float)sv[r];
      co[r] = (_Float16)(g * s * (1.0f - s));
      uo[r] = (_Float16)(g * s);
    }
    *(v8h*)(cc + smp * 256 + mb) = co;
    *(v8h*)(uout + smp * 256 + mb) = uo;
  }
}

// ---------------- tangent forward ----------------
__device__ __forceinline__ void tan_gemm_first(const _Float16* __restrict__ W0h,
                                               const _Float16* __restrict__ V,
                                               _Float16* __restrict__ Tout, int wave, int lane) {
  for (int t = wave; t < 256; t += 8) {
    int m0 = (t >> 4) << 4, n0 = (t & 15) << 4;
    v8f acc = {};
    v16h a = loadA(W0h, 32, m0, 0, lane);
    v16h b = loadB(V, 32, n0, 0, lane);
    acc = WMMA16(a, b, acc);
    int col = n0 + (lane & 15);
    int mb  = m0 + (lane >> 4) * 8;
    v8h o;
#pragma unroll
    for (int r = 0; r < 8; r++) o[r] = (_Float16)acc[r];
    *(v8h*)(Tout + (size_t)col * 256 + mb) = o;
  }
}

// Tout = W * (sigma ⊙ Tin)  -- diag scaling fused into B-operand load
__device__ __forceinline__ void tan_gemm(const _Float16* __restrict__ W,
                                         const _Float16* __restrict__ Tin,
                                         const _Float16* __restrict__ sg,
                                         _Float16* __restrict__ Tout, int wave, int lane) {
  for (int t = wave; t < 256; t += 8) {
    int m0 = (t >> 4) << 4, n0 = (t & 15) << 4;
    int col = n0 + (lane & 15);
    int smp = col >> 4;
    int h16 = (lane >> 4) * 16;
    v8f acc = {};
#pragma unroll
    for (int kb = 0; kb < 256; kb += 32) {
      v16h a = loadA(W, 256, m0, kb, lane);
      v16h b = *(const v16h*)(Tin + (size_t)col * 256 + kb + h16);
      v16h s = *(const v16h*)(sg + smp * 256 + kb + h16);
      b = b * s;
      acc = WMMA16(a, b, acc);
    }
    int mb = m0 + (lane >> 4) * 8;
    v8h o;
#pragma unroll
    for (int r = 0; r < 8; r++) o[r] = (_Float16)acc[r];
    *(v8h*)(Tout + (size_t)col * 256 + mb) = o;
  }
}

// ---------------- weighted backward: S_k = c_k⊙T_k + sigma_k⊙(W_k^T S_{k+1}) ----
// FIRST: S4 is virtual = c4⊙T4, except grad column (j==13) which is u4.
template <bool FIRST>
__device__ __forceinline__ void bwd_gemm(const _Float16* __restrict__ Wt,
                                         const _Float16* __restrict__ Bsrc,
                                         const _Float16* __restrict__ c_next,
                                         const _Float16* __restrict__ u4,
                                         const _Float16* __restrict__ Tk,
                                         const _Float16* __restrict__ ck,
                                         const _Float16* __restrict__ sk,
                                         _Float16* __restrict__ Sout, int wave, int lane) {
  for (int t = wave; t < 256; t += 8) {
    int m0 = (t >> 4) << 4, n0 = (t & 15) << 4;
    int col = n0 + (lane & 15);
    int smp = col >> 4, j = col & 15;
    int h16 = (lane >> 4) * 16;
    v8f acc = {};
#pragma unroll
    for (int kb = 0; kb < 256; kb += 32) {
      v16h a = loadA(Wt, 256, m0, kb, lane);
      v16h b;
      if (FIRST) {
        if (j == 13) {
          b = *(const v16h*)(u4 + smp * 256 + kb + h16);
        } else {
          b = *(const v16h*)(Bsrc + (size_t)col * 256 + kb + h16);
          b = b * *(const v16h*)(c_next + smp * 256 + kb + h16);
        }
      } else {
        b = *(const v16h*)(Bsrc + (size_t)col * 256 + kb + h16);
      }
      acc = WMMA16(a, b, acc);
    }
    int mb = m0 + (lane >> 4) * 8;
    v8h tk = *(const v8h*)(Tk + (size_t)col * 256 + mb);
    v8h cv = *(const v8h*)(ck + smp * 256 + mb);
    v8h sv = *(const v8h*)(sk + smp * 256 + mb);
    v8h o;
#pragma unroll
    for (int r = 0; r < 8; r++)
      o[r] = (_Float16)((float)cv[r] * (float)tk[r] + (float)sv[r] * acc[r]);
    *(v8h*)(Sout + (size_t)col * 256 + mb) = o;
  }
}

// ---------------- final projection: HV = W0^T S1  (M=32 padded, f32 out) ----
__device__ __forceinline__ void hv_gemm(const _Float16* __restrict__ Wt0,
                                        const _Float16* __restrict__ S1,
                                        float* __restrict__ HV, int wave, int lane) {
  for (int t = wave; t < 32; t += 8) {
    int m0 = (t >> 4) << 4, n0 = (t & 15) << 4;
    v8f acc = {};
#pragma unroll
    for (int kb = 0; kb < 256; kb += 32) {
      v16h a = loadA(Wt0, 256, m0, kb, lane);
      v16h b = loadB(S1, 256, n0, kb, lane);
      acc = WMMA16(a, b, acc);
    }
    int col = n0 + (lane & 15);
    int mb  = m0 + (lane >> 4) * 8;
    *(v8f*)(HV + (size_t)col * 32 + mb) = acc;   // 32B-aligned vector store
  }
}

// ---------------- prologue: f32 weights -> f16 (+transposes, padding) ----------------
__global__ __launch_bounds__(256) void lnn_prep(const float* __restrict__ W0,
                                                const float* __restrict__ W1,
                                                const float* __restrict__ W2,
                                                const float* __restrict__ W3,
                                                const float* __restrict__ W4,
                                                _Float16* __restrict__ ws) {
  int tid = blockIdx.x * 256 + threadIdx.x;  // 65536 threads
  int i = tid >> 8, j = tid & 255;
  const float* Ws[3] = {W1, W2, W3};
#pragma unroll
  for (int k = 0; k < 3; k++) {
    float w = Ws[k][i * 256 + j];
    ws[S_WH(k) + (size_t)i * 256 + j] = (_Float16)w;
    ws[S_WT(k) + (size_t)j * 256 + i] = (_Float16)w;
  }
  if (j < 32) ws[S_WH0 + (size_t)i * 32 + j] = (j < 24) ? (_Float16)W0[i * 24 + j] : (_Float16)0.0f;
  if (i < 32) ws[S_WT0 + (size_t)i * 256 + j] = (i < 24) ? (_Float16)W0[j * 24 + i] : (_Float16)0.0f;
  if (i == 0) ws[S_W4 + j] = (_Float16)W4[j];
}

// ---------------- main persistent kernel ----------------
__global__ __launch_bounds__(256) void lnn_main(const float* __restrict__ q,
                                                const float* __restrict__ qdot,
                                                const float* __restrict__ b0,
                                                const float* __restrict__ b1,
                                                const float* __restrict__ b2,
                                                const float* __restrict__ b3,
                                                char* __restrict__ ws, float* __restrict__ out,
                                                int Bn, int ntiles) {
  const int tid = threadIdx.x, wave = tid >> 5, lane = tid & 31;
  _Float16* st = (_Float16*)ws;
  char* blk = ws + S_BYTES + (size_t)blockIdx.x * B_BYTES;
  _Float16* bh = (_Float16*)blk;
  float* HV = (float*)(blk + B_HVOFF);

  for (int t = blockIdx.x; t < ntiles; t += gridDim.x) {
    int sb = t * NS;

    // build input tangents V [col][32] and network input z0 [sample][32]
    {
      int c = tid, s = c >> 4, j = c & 15;
      bool ok = (sb + s) < Bn;
      _Float16* Vc = bh + B_V + (size_t)c * 32;
      for (int r = 0; r < 32; r++) {
        float v = 0.0f;
        if (ok) {
          if (j < NDOF)      v = (r == NDOF + j) ? 1.0f : 0.0f;
          else if (j == 12 && r < NDOF) v = qdot[(size_t)(sb + s) * NDOF + r];
        }
        Vc[r] = (_Float16)v;
      }
      if (tid < NS) {
        int s2 = tid;
        bool ok2 = (sb + s2) < Bn;
        _Float16* z0 = bh + B_Z0 + (size_t)s2 * 32;
        for (int r = 0; r < 32; r++) {
          float v = 0.0f;
          if (ok2) {
            if (r < NDOF)        v = q[(size_t)(sb + s2) * NDOF + r];
            else if (r < 2*NDOF) v = qdot[(size_t)(sb + s2) * NDOF + (r - NDOF)];
          }
          z0[r] = (_Float16)v;
        }
      }
    }
    gsync();

    // ---- forward activations: sigma_k (and z_k), last layer -> c4, u4 ----
    fwd_gemm<32, 32, 32>(st + S_WH0, bh + B_Z0, b0, bh + B_SIG(0), bh + B_ZA,
                         nullptr, nullptr, nullptr, wave, lane);
    gsync();
    fwd_gemm<256, 256, 256>(st + S_WH(0), bh + B_ZA, b1, bh + B_SIG(1), bh + B_ZB,
                            nullptr, nullptr, nullptr, wave, lane);
    gsync();
    fwd_gemm<256, 256, 256>(st + S_WH(1), bh + B_ZB, b2, bh + B_SIG(2), bh + B_ZA,
                            nullptr, nullptr, nullptr, wave, lane);
    gsync();
    fwd_gemm<256, 256, 256>(st + S_WH(2), bh + B_ZA, b3, bh + B_SIG(3), nullptr,
                            st + S_W4, bh + B_CC(3), bh + B_UA, wave, lane);
    gsync();

    // ---- gradient backward: c3, c2, c1 (u4 preserved in UA) ----
    grad_gemm(st + S_WT(2), bh + B_UA, bh + B_SIG(2), bh + B_CC(2), bh + B_UB, wave, lane);
    gsync();
    grad_gemm(st + S_WT(1), bh + B_UB, bh + B_SIG(1), bh + B_CC(1), bh + B_UC, wave, lane);
    gsync();
    grad_gemm(st + S_WT(0), bh + B_UC, bh + B_SIG(0), bh + B_CC(0), bh + B_UB, wave, lane);
    gsync();

    // ---- tangent forward: T1..T4 ----
    tan_gemm_first(st + S_WH0, bh + B_V, bh + B_T(0), wave, lane);
    gsync();
    tan_gemm(st + S_WH(0), bh + B_T(0), bh + B_SIG(0), bh + B_T(1), wave, lane);
    gsync();
    tan_gemm(st + S_WH(1), bh + B_T(1), bh + B_SIG(1), bh + B_T(2), wave, lane);
    gsync();
    tan_gemm(st + S_WH(2), bh + B_T(2), bh + B_SIG(2), bh + B_T(3), wave, lane);
    gsync();

    // ---- weighted backward: S3 -> SE, S2 -> T4buf, S1 -> SE ----
    bwd_gemm<true>(st + S_WT(2), bh + B_T(3), bh + B_CC(3), bh + B_UA,
                   bh + B_T(2), bh + B_CC(2), bh + B_SIG(2), bh + B_SE, wave, lane);
    gsync();
    bwd_gemm<false>(st + S_WT(1), bh + B_SE, nullptr, nullptr,
                    bh + B_T(1), bh + B_CC(1), bh + B_SIG(1), bh + B_T(3), wave, lane);
    gsync();
    bwd_gemm<false>(st + S_WT(0), bh + B_T(3), nullptr, nullptr,
                    bh + B_T(0), bh + B_CC(0), bh + B_SIG(0), bh + B_SE, wave, lane);
    gsync();
    hv_gemm(st + S_WT0, bh + B_SE, HV, wave, lane);
    gsync();

    // ---- per-sample 12x12 solve: (M + 0.01 I) qddot = g_q - C qdot ----
    if (tid < NS && (sb + tid) < Bn) {
      int s = tid;
      const float* base = HV + (size_t)(s * 16) * 32;
      float A[12][12], rhs[12], x[12];
      for (int i = 0; i < 12; i++)
        for (int j = 0; j < 12; j++) A[i][j] = base[(size_t)j * 32 + 12 + i];
      for (int i = 0; i < 12; i++) {
        A[i][i] += 0.01f;
        rhs[i] = base[(size_t)13 * 32 + i] - base[(size_t)12 * 32 + 12 + i];
      }
      for (int k = 0; k < 12; k++) {
        int p = k; float mx = fabsf(A[k][k]);
        for (int r = k + 1; r < 12; r++) { float v = fabsf(A[r][k]); if (v > mx) { mx = v; p = r; } }
        if (p != k) {
          for (int c = k; c < 12; c++) { float tmp = A[k][c]; A[k][c] = A[p][c]; A[p][c] = tmp; }
          float tb = rhs[k]; rhs[k] = rhs[p]; rhs[p] = tb;
        }
        float inv = 1.0f / A[k][k];
        for (int r = k + 1; r < 12; r++) {
          float f = A[r][k] * inv;
          for (int c = k; c < 12; c++) A[r][c] -= f * A[k][c];
          rhs[r] -= f * rhs[k];
        }
      }
      for (int i = 11; i >= 0; i--) {
        float acc = rhs[i];
        for (int c = i + 1; c < 12; c++) acc -= A[i][c] * x[c];
        x[i] = acc / A[i][i];
      }
      for (int i = 0; i < 12; i++) out[(size_t)(sb + s) * NDOF + i] = x[i];
    }
    gsync();  // buffers reused next tile
  }
}

extern "C" void kernel_launch(void* const* d_in, const int* in_sizes, int n_in,
                              void* d_out, int out_size, void* d_ws, size_t ws_size,
                              hipStream_t stream) {
  (void)n_in; (void)out_size;
  const float* q  = (const float*)d_in[0];
  const float* qd = (const float*)d_in[1];
  const float* W0 = (const float*)d_in[2];
  const float* b0 = (const float*)d_in[3];
  const float* W1 = (const float*)d_in[4];
  const float* b1 = (const float*)d_in[5];
  const float* W2 = (const float*)d_in[6];
  const float* b2 = (const float*)d_in[7];
  const float* W3 = (const float*)d_in[8];
  const float* b3 = (const float*)d_in[9];
  const float* W4 = (const float*)d_in[10];

  int Bn = in_sizes[0] / NDOF;
  int ntiles = (Bn + NS - 1) / NS;

  lnn_prep<<<256, 256, 0, stream>>>(W0, W1, W2, W3, W4, (_Float16*)d_ws);

  size_t avail = (ws_size > S_BYTES) ? (ws_size - S_BYTES) : 0;
  int nblk = (int)(avail / B_BYTES);
  if (nblk < 1) nblk = 1;
  if (nblk > ntiles) nblk = ntiles;

  lnn_main<<<nblk, 256, 0, stream>>>(q, qd, b0, b1, b2, b3,
                                     (char*)d_ws, (float*)d_out, Bn, ntiles);
}
